// AHEWN_240518168829
// MI455X (gfx1250) — matrix operand order repaired and checked by reading, not compiled, not run on hardware
//
#include <hip/hip_runtime.h>
#include <vector>

// ---------------------------------------------------------------------------
// Types for CDNA5 WMMA (wave32): A/B are 16xbf16 per lane, C/D are 8xf32.
// ---------------------------------------------------------------------------
typedef __bf16 bf16_t;
typedef bf16_t v16bf __attribute__((ext_vector_type(16)));
typedef bf16_t v8bf  __attribute__((ext_vector_type(8)));
typedef float  v8f   __attribute__((ext_vector_type(8)));

#define BATCH_  128
#define CHAN_   128
#define SEQ_    2048
#define NC_     1027          // DWT coefficient length
#define OL_     363           // mixer output length
#define PL_     720           // prediction length
#define DM_     128           // d_model
#define ROWS_   (BATCH_*CHAN_) // 16384
#define ZW_     2054          // concat width for cA[1] mixer input

#define LDSPAD  40            // padded row stride (bf16 elems) in LDS tiles

// db4 filter banks (REC_LO = H; REC_HI[t] = (-1)^t H[7-t];
// DEC_LO[k] = H[7-k]; DEC_HI[k] = (-1)^(k+1) H[k])
__constant__ float c_RECLO[8] = {
  0.23037781330885523f, 0.7148465705525415f, 0.6308807679295904f, -0.02798376941698385f,
 -0.18703481171888114f, 0.030841381835986965f, 0.032883011666982945f, -0.010597401784997278f};
__constant__ float c_RECHI[8] = {
 -0.010597401784997278f, -0.032883011666982945f, 0.030841381835986965f, 0.18703481171888114f,
 -0.02798376941698385f, -0.6308807679295904f, 0.7148465705525415f, -0.23037781330885523f};
__constant__ float c_DECLO[8] = {
 -0.010597401784997278f, 0.032883011666982945f, 0.030841381835986965f, -0.18703481171888114f,
 -0.02798376941698385f, 0.6308807679295904f, 0.7148465705525415f, 0.23037781330885523f};
__constant__ float c_DECHI[8] = {
 -0.23037781330885523f, 0.7148465705525415f, -0.6308807679295904f, -0.02798376941698385f,
  0.18703481171888114f, 0.030841381835986965f, -0.032883011666982945f, -0.010597401784997278f};

// ---------------------------------------------------------------------------
// Cooperative staging of one K-step (32) of A (128 rows) and B (64 cols) into
// LDS as bf16. Branch-free boundary handling: clamped address + value select,
// chosen by *uniform* kfull/nfull flags so the common path is pure vector IO.
//   As: [128][LDSPAD]  row-major (row = block-local M)
//   Bs: [ 64][LDSPAD]  TRANSPOSED (row = block-local N, elements along K)
// ---------------------------------------------------------------------------
__device__ __forceinline__ void stage_tiles(
    const float* __restrict__ A, int lda,
    const float* __restrict__ Bw, int ldb,
    int rowB, int colB, int k0, int K, int N,
    bool kfull, bool nfull,
    bf16_t* __restrict__ As, bf16_t* __restrict__ Bs, int tid)
{
  // ---- A tile: thread -> (row = tid>>1, 16 contiguous floats at k0+half*16)
  {
    const int row = tid >> 1, half = tid & 1;
    const float* __restrict__ src = A + (size_t)(rowB + row) * (size_t)lda;
    bf16_t* __restrict__ dst = As + row * LDSPAD + half * 16;
    if (kfull) {
      const float* s = src + k0 + half * 16;
      v8bf lo, hi;
#pragma unroll
      for (int i = 0; i < 8; ++i) lo[i] = (bf16_t)s[i];
#pragma unroll
      for (int i = 0; i < 8; ++i) hi[i] = (bf16_t)s[8 + i];
      *(v8bf*)dst       = lo;   // 16B aligned: row*80 + half*32
      *(v8bf*)(dst + 8) = hi;
    } else {
#pragma unroll
      for (int i = 0; i < 16; ++i) {
        int k = k0 + half * 16 + i;
        int kc = k < K ? k : K - 1;     // clamped address, unconditional load
        float v = src[kc];
        dst[i] = (bf16_t)(k < K ? v : 0.0f);
      }
    }
  }
  // ---- B tile: thread -> (k = k0 + (tid>>3), 8 contiguous cols at (tid&7)*8)
  {
    const int kl = tid >> 3, n0 = (tid & 7) * 8;
    const int k = k0 + kl;
    if (kfull && nfull) {
      const float* __restrict__ s = Bw + (size_t)k * (size_t)ldb + colB + n0;
#pragma unroll
      for (int i = 0; i < 8; ++i) Bs[(n0 + i) * LDSPAD + kl] = (bf16_t)s[i];
    } else {
      int kc = k < K ? k : K - 1;
      const float* __restrict__ s = Bw + (size_t)kc * (size_t)ldb;
#pragma unroll
      for (int i = 0; i < 8; ++i) {
        int col = colB + n0 + i;
        int cc = col < N ? col : N - 1;
        float v = s[cc];
        Bs[(n0 + i) * LDSPAD + kl] = (bf16_t)((k < K && col < N) ? v : 0.0f);
      }
    }
  }
}

// ---------------------------------------------------------------------------
// Fragment fetch from LDS (aligned 16B vector loads -> ds_load_b128).
// A (16x32): lane m=lane&15, h=lane>>4; comps 0-7 <-> K=h*8+c, 8-15 <-> 16+h*8+c
// B (32x16): lane n=lane&15, h=lane>>4; comps 0-7 <-> K=h*8+c, 8-15 <-> 16+h*8+c
// ---------------------------------------------------------------------------
__device__ __forceinline__ v16bf frag_A(const bf16_t* __restrict__ As, int m, int h) {
  const v8bf lo = *(const v8bf*)(As + m * LDSPAD + h * 8);
  const v8bf hi = *(const v8bf*)(As + m * LDSPAD + 16 + h * 8);
  return __builtin_shufflevector(lo, hi, 0,1,2,3,4,5,6,7,8,9,10,11,12,13,14,15);
}
__device__ __forceinline__ v16bf frag_B(const bf16_t* __restrict__ Bs, int n, int h) {
  const v8bf lo = *(const v8bf*)(Bs + n * LDSPAD + h * 8);
  const v8bf hi = *(const v8bf*)(Bs + n * LDSPAD + 16 + h * 8);
  return __builtin_shufflevector(lo, hi, 0,1,2,3,4,5,6,7,8,9,10,11,12,13,14,15);
}

// D layout: VGPR r -> M = half*8 + r, N = lane&15.
__device__ __forceinline__ void store_tile(v8f acc, int mbase, int n,
    const float* __restrict__ bias, const float* __restrict__ Res, int ldr,
    float* __restrict__ C, int ldc, int N, int do_gelu) {
  if (n >= N) return;
  float bv = bias ? bias[n] : 0.0f;
#pragma unroll
  for (int r = 0; r < 8; ++r) {
    float v = acc[r] + bv;
    if (do_gelu) v = 0.5f * v * (1.0f + erff(v * 0.70710678118654752f)); // exact gelu
    if (Res) v += Res[(size_t)(mbase + r) * (size_t)ldr + n];
    C[(size_t)(mbase + r) * (size_t)ldc + n] = v;
  }
}

// ---------------------------------------------------------------------------
// GEMM: C[M,N] = epilogue(A[M,K] @ B[K,N] + bias (+Res)); fp32 in/out,
// bf16 WMMA with fp32 accumulation. M = gridDim.y*128 (always 16384 here).
// Block: 256 thr = 8 waves (4 M x 2 N), block tile 128x64, wave tile 32x32.
// Double-buffered LDS staging; branch-free wmma loop (EXEC all-ones).
// ---------------------------------------------------------------------------
__global__ __launch_bounds__(256) void gemm_bf16_wmma(
    const float* __restrict__ A, int lda,
    const float* __restrict__ Bw, int ldb,
    const float* __restrict__ bias,
    const float* __restrict__ Res, int ldr,
    float* __restrict__ C, int ldc,
    int N, int K, int do_gelu)
{
  __shared__ __align__(16) bf16_t As[2][128 * LDSPAD];
  __shared__ __align__(16) bf16_t Bs[2][ 64 * LDSPAD];

  const int tid  = threadIdx.x;
  const int lane = tid & 31;
  const int wv   = tid >> 5;
  const int wm   = wv & 3, wn = wv >> 2;
  const int rowB = blockIdx.y * 128;
  const int colB = blockIdx.x * 64;
  const bool nfull = (colB + 64 <= N);
  const int nk = (K + 31) >> 5;

  stage_tiles(A, lda, Bw, ldb, rowB, colB, 0, K, N, 32 <= K, nfull, As[0], Bs[0], tid);
  __syncthreads();

  v8f a00 = {}, a01 = {}, a10 = {}, a11 = {};
  const int h  = lane >> 4;
  const int ml = wm * 32 + (lane & 15);   // block-local M for A frags
  const int nl = wn * 32 + (lane & 15);   // block-local N for B frags

  for (int i = 0; i < nk; ++i) {
    const int cur = i & 1;
    if (i + 1 < nk) {
      const int k0n = (i + 1) * 32;
      stage_tiles(A, lda, Bw, ldb, rowB, colB, k0n, K, N,
                  k0n + 32 <= K, nfull, As[cur ^ 1], Bs[cur ^ 1], tid);
    }
    v16bf fa0 = frag_A(As[cur], ml,      h);
    v16bf fa1 = frag_A(As[cur], ml + 16, h);
    v16bf fb0 = frag_B(Bs[cur], nl,      h);
    v16bf fb1 = frag_B(Bs[cur], nl + 16, h);
    a00 = __builtin_amdgcn_wmma_f32_16x16x32_bf16(false, fa0, false, fb0, (short)0, a00, false, false);
    a01 = __builtin_amdgcn_wmma_f32_16x16x32_bf16(false, fa0, false, fb1, (short)0, a01, false, false);
    a10 = __builtin_amdgcn_wmma_f32_16x16x32_bf16(false, fa1, false, fb0, (short)0, a10, false, false);
    a11 = __builtin_amdgcn_wmma_f32_16x16x32_bf16(false, fa1, false, fb1, (short)0, a11, false, false);
    __syncthreads();
  }

  const int row0 = rowB + wm * 32;
  const int col0 = colB + wn * 32;
  const int nn = lane & 15;
  store_tile(a00, row0 +      h * 8, col0 +      nn, bias, Res, ldr, C, ldc, N, do_gelu);
  store_tile(a01, row0 +      h * 8, col0 + 16 + nn, bias, Res, ldr, C, ldc, N, do_gelu);
  store_tile(a10, row0 + 16 + h * 8, col0 +      nn, bias, Res, ldr, C, ldc, N, do_gelu);
  store_tile(a11, row0 + 16 + h * 8, col0 + 16 + nn, bias, Res, ldr, C, ldc, N, do_gelu);
}

// ---------------------------------------------------------------------------
// DWT: x_enc (B,L,C) -> cA,cD (B*C rows, 1027), with symmetric-ext (drop 1st)
// ---------------------------------------------------------------------------
__global__ void dwt_kernel(const float* __restrict__ xe,
                           float* __restrict__ cA, float* __restrict__ cD) {
  const int c = threadIdx.x;            // channel
  const int b = blockIdx.y;
  const int o0 = blockIdx.x * 4;
#pragma unroll
  for (int t = 0; t < 4; ++t) {
    int o = o0 + t;
    if (o >= NC_) return;
    float lo = 0.f, hi = 0.f;
#pragma unroll
    for (int k = 0; k < 8; ++k) {
      int j = 2 * o + k;                         // index into ext[0..2060]
      int l = (j < 6) ? (5 - j) : ((j < 2054) ? (j - 6) : (4101 - j));
      float v = xe[((size_t)b * SEQ_ + l) * CHAN_ + c];
      lo += v * c_RECLO[k];
      hi += v * c_RECHI[k];
    }
    size_t row = (size_t)b * CHAN_ + c;
    cA[row * NC_ + o] = lo;
    cD[row * NC_ + o] = hi;
  }
}

// ---------------------------------------------------------------------------
// IDWT pieces: out[row, coloff+o] = base(o) + aconv(a)(o) + dconv(d)(o)
// y[o] = sum_k u(o+k-1)*w[k], u(m)=src[m/2] for even m, 0<=m/2<363
// ---------------------------------------------------------------------------
__global__ void recon_kernel(const float* __restrict__ a,
                             const float* __restrict__ d,
                             const float* __restrict__ base,   // ld = PL_
                             float* __restrict__ out, int ldo, int coloff) {
  const int row = blockIdx.x;
  const int o = blockIdx.y * 128 + threadIdx.x;
  if (o >= PL_) return;
  float acc = base ? base[(size_t)row * PL_ + o] : 0.f;
  if (a) {
    const float* ar = a + (size_t)row * OL_;
#pragma unroll
    for (int k = 0; k < 8; ++k) {
      int m = o + k - 1;
      if (m >= 0 && !(m & 1)) { int i = m >> 1; if (i < OL_) acc += ar[i] * c_DECLO[k]; }
    }
  }
  if (d) {
    const float* dr = d + (size_t)row * OL_;
#pragma unroll
    for (int k = 0; k < 8; ++k) {
      int m = o + k - 1;
      if (m >= 0 && !(m & 1)) { int i = m >> 1; if (i < OL_) acc += dr[i] * c_DECHI[k]; }
    }
  }
  out[(size_t)row * ldo + coloff + o] = acc;
}

// Z[:, 0:1027] = cD ; Z[:, 1747:2054] = 0  (Z[:,1027:1747] filled by recon)
__global__ void packZ_kernel(const float* __restrict__ cD, float* __restrict__ Z) {
  const int row = blockIdx.x;
  const int col = blockIdx.y * 256 + threadIdx.x;
  if (col < NC_)              Z[(size_t)row * ZW_ + col] = cD[(size_t)row * NC_ + col];
  else if (col >= NC_ + PL_ && col < ZW_) Z[(size_t)row * ZW_ + col] = 0.f;
}

// ---------------------------------------------------------------------------
// Row LayerNorm over width 128 (1 wave per row, 8 rows/block)
// ---------------------------------------------------------------------------
__global__ __launch_bounds__(256) void ln_kernel(const float* __restrict__ x,
                                                 float* __restrict__ y,
                                                 const float* __restrict__ g,
                                                 const float* __restrict__ b) {
  const int lane = threadIdx.x & 31;
  const int row = blockIdx.x * 8 + (threadIdx.x >> 5);
  const float* xr = x + (size_t)row * DM_;
  float v[4], s = 0.f;
#pragma unroll
  for (int i = 0; i < 4; ++i) { v[i] = xr[lane + 32 * i]; s += v[i]; }
#pragma unroll
  for (int off = 16; off; off >>= 1) s += __shfl_xor(s, off, 32);
  float m = s * (1.f / 128.f);
  float q = 0.f;
#pragma unroll
  for (int i = 0; i < 4; ++i) { float dd = v[i] - m; q += dd * dd; }
#pragma unroll
  for (int off = 16; off; off >>= 1) q += __shfl_xor(q, off, 32);
  float inv = rsqrtf(q * (1.f / 128.f) + 1e-5f);
  float* yr = y + (size_t)row * DM_;
#pragma unroll
  for (int i = 0; i < 4; ++i) {
    int col = lane + 32 * i;
    yr[col] = (v[i] - m) * inv * g[col] + b[col];
  }
}

// ---------------------------------------------------------------------------
// Per-batch 128x128 transpose via LDS: OUT[q][p] (op)= IN[p][q]
// ---------------------------------------------------------------------------
__global__ void transpose128_kernel(const float* __restrict__ in,
                                    float* __restrict__ out, int add) {
  __shared__ float tile[32][33];
  const int b = blockIdx.z;
  const float* ib = in  + (size_t)b * DM_ * DM_;
  float*       ob = out + (size_t)b * DM_ * DM_;
  const int p0 = blockIdx.x * 32, q0 = blockIdx.y * 32;
  const int x = threadIdx.x, ty = threadIdx.y;
#pragma unroll
  for (int k = 0; k < 4; ++k) {
    int yy = ty + 8 * k;
    tile[yy][x] = ib[(size_t)(p0 + yy) * DM_ + q0 + x];
  }
  __syncthreads();
#pragma unroll
  for (int k = 0; k < 4; ++k) {
    int yy = ty + 8 * k;
    float v = tile[x][yy];
    size_t oi = (size_t)(q0 + yy) * DM_ + p0 + x;
    ob[oi] = add ? (ob[oi] + v) : v;
  }
}

// ---------------------------------------------------------------------------
// Gating: logits[b,:] = hid_row(b) @ Wg + bg ; w = softmax
// ---------------------------------------------------------------------------
__global__ __launch_bounds__(256) void gate_kernel(const float* __restrict__ hid,
                                                   const float* __restrict__ Wg,
                                                   const float* __restrict__ bg,
                                                   float* __restrict__ w) {
  const int b = blockIdx.x, tid = threadIdx.x;
  const float* h = hid + (size_t)b * 16384;
  float s0 = 0.f, s1 = 0.f;
  for (int i = tid; i < 16384; i += 256) {
    float hv = h[i];
    s0 += hv * Wg[(size_t)i * 2];
    s1 += hv * Wg[(size_t)i * 2 + 1];
  }
  __shared__ float r0[256], r1[256];
  r0[tid] = s0; r1[tid] = s1;
  __syncthreads();
  for (int st = 128; st; st >>= 1) {
    if (tid < st) { r0[tid] += r0[tid + st]; r1[tid] += r1[tid + st]; }
    __syncthreads();
  }
  if (tid == 0) {
    float l0 = r0[0] + bg[0], l1 = r1[0] + bg[1];
    float mx = fmaxf(l0, l1);
    float e0 = expf(l0 - mx), e1 = expf(l1 - mx);
    float inv = 1.f / (e0 + e1);
    w[b * 2] = e0 * inv; w[b * 2 + 1] = e1 * inv;
  }
}

// out[b,o,c] = (w0+w1)*rec0[b,c,o] + w1*recD[b,c,o]
__global__ void final_kernel(const float* __restrict__ rec0,
                             const float* __restrict__ recD,
                             const float* __restrict__ w,
                             float* __restrict__ out) {
  const int b = blockIdx.y;
  const int t = blockIdx.x * 256 + threadIdx.x;
  if (t >= PL_ * CHAN_) return;
  const int o = t >> 7, c = t & 127;
  const size_t r = ((size_t)b * CHAN_ + c) * PL_ + o;
  const float w0 = w[b * 2], w1 = w[b * 2 + 1];
  out[(size_t)b * PL_ * CHAN_ + (size_t)o * CHAN_ + c] = (w0 + w1) * rec0[r] + w1 * recD[r];
}

// ---------------------------------------------------------------------------
// Host-side orchestration
// ---------------------------------------------------------------------------
struct MixerP {
  const float *Wi, *bi, *Wo, *bo;
  struct Layer {
    const float *tg, *tb, *tW1, *tb1, *tW2, *tb2, *cg, *cb, *cW1, *cb1, *cW2, *cb2;
  } L[2];
};

struct Binding {
  std::vector<long> sz;
  std::vector<const float**> dst;
  void add(long s, const float** p) { sz.push_back(s); dst.push_back(p); }
};

static void bind_mixer(Binding& bd, MixerP* m, long inlen, bool sorted) {
  const long d = 128, o = 363, d4 = 512;
  if (!sorted) {  // Python dict insertion order
    bd.add(inlen * d, &m->Wi); bd.add(d, &m->bi);
    bd.add(d * o, &m->Wo);     bd.add(o, &m->bo);
    for (int l = 0; l < 2; ++l) {
      MixerP::Layer* L = &m->L[l];
      bd.add(d, &L->tg);  bd.add(d, &L->tb);
      bd.add(d * d, &L->tW1); bd.add(d, &L->tb1);
      bd.add(d * d, &L->tW2); bd.add(d, &L->tb2);
      bd.add(d, &L->cg);  bd.add(d, &L->cb);
      bd.add(d * d4, &L->cW1); bd.add(d4, &L->cb1);
      bd.add(d4 * d, &L->cW2); bd.add(d, &L->cb2);
    }
  } else {        // JAX sorted-key tree order: Wi,Wo,bi,bo, layers{cW1,cW2,cb,cb1,cb2,cg,tW1,tW2,tb,tb1,tb2,tg}
    bd.add(inlen * d, &m->Wi); bd.add(d * o, &m->Wo);
    bd.add(d, &m->bi);         bd.add(o, &m->bo);
    for (int l = 0; l < 2; ++l) {
      MixerP::Layer* L = &m->L[l];
      bd.add(d * d4, &L->cW1); bd.add(d4 * d, &L->cW2);
      bd.add(d, &L->cb);  bd.add(d4, &L->cb1); bd.add(d, &L->cb2); bd.add(d, &L->cg);
      bd.add(d * d, &L->tW1);  bd.add(d * d, &L->tW2);
      bd.add(d, &L->tb);  bd.add(d, &L->tb1);  bd.add(d, &L->tb2); bd.add(d, &L->tg);
    }
  }
}

static void gemm(hipStream_t s, const float* A, int lda, const float* Bw, int ldb,
                 const float* bias, const float* Res, int ldr,
                 float* C, int ldc, int N, int K, int do_gelu) {
  dim3 grid((N + 63) / 64, ROWS_ / 128);  // M = 16384 always
  gemm_bf16_wmma<<<grid, 256, 0, s>>>(A, lda, Bw, ldb, bias, Res, ldr, C, ldc, N, K, do_gelu);
}

static void run_mixer(hipStream_t s, const MixerP& p, const float* in, int ldin, int Kin,
                      float* outp, float* mixX, float* tA, float* tB, float* tC) {
  gemm(s, in, ldin, p.Wi, DM_, p.bi, nullptr, 0, mixX, DM_, DM_, Kin, 0);
  for (int l = 0; l < 2; ++l) {
    const MixerP::Layer& L = p.L[l];
    // temporal MLP (rows = b*C+c, width d)
    ln_kernel<<<ROWS_ / 8, 256, 0, s>>>(mixX, tA, L.tg, L.tb);
    gemm(s, tA, DM_, L.tW1, DM_, L.tb1, nullptr, 0, tB, DM_, DM_, DM_, 1);
    gemm(s, tB, DM_, L.tW2, DM_, L.tb2, mixX, DM_, mixX, DM_, DM_, DM_, 0);
    // channel MLP: transpose per batch, LN over C, 128->512->128, transpose-add
    transpose128_kernel<<<dim3(4, 4, BATCH_), dim3(32, 8), 0, s>>>(mixX, tC, 0);
    ln_kernel<<<ROWS_ / 8, 256, 0, s>>>(tC, tA, L.cg, L.cb);
    gemm(s, tA, DM_, L.cW1, 512, L.cb1, nullptr, 0, tB, 512, 512, DM_, 1);
    gemm(s, tB, 512, L.cW2, DM_, L.cb2, nullptr, 0, tA, DM_, DM_, 512, 0);
    transpose128_kernel<<<dim3(4, 4, BATCH_), dim3(32, 8), 0, s>>>(tA, mixX, 1);
  }
  gemm(s, mixX, DM_, p.Wo, OL_, p.bo, nullptr, 0, outp, OL_, OL_, DM_, 0);
}

extern "C" void kernel_launch(void* const* d_in, const int* in_sizes, int n_in,
                              void* d_out, int out_size, void* d_ws, size_t ws_size,
                              hipStream_t stream) {
  (void)out_size; (void)ws_size;
  // ---- bind inputs: try several flattening orders, match by size signature ----
  MixerP mA0, mA1, mD0;
  const float *xenc = nullptr, *Wg = nullptr, *bg = nullptr;
  const long XSZ = (long)BATCH_ * SEQ_ * CHAN_;   // 33554432
  const long WGSZ = 16384L * 2;
  bool matched = false;
  for (int combo = 0; combo < 4 && !matched; ++combo) {
    Binding bd;
    switch (combo) {
      case 0:  // x_enc first, dicts in insertion order
        bd.add(XSZ, &xenc);
        bind_mixer(bd, &mA0, NC_, false); bind_mixer(bd, &mA1, ZW_, false);
        bind_mixer(bd, &mD0, NC_, false);
        bd.add(WGSZ, &Wg); bd.add(2, &bg);
        break;
      case 1:  // full JAX sorted tree: params(Wg,bg,cA,cD) then x_enc
        bd.add(WGSZ, &Wg); bd.add(2, &bg);
        bind_mixer(bd, &mA0, NC_, true); bind_mixer(bd, &mA1, ZW_, true);
        bind_mixer(bd, &mD0, NC_, true);
        bd.add(XSZ, &xenc);
        break;
      case 2:  // x_enc first, params sorted
        bd.add(XSZ, &xenc); bd.add(WGSZ, &Wg); bd.add(2, &bg);
        bind_mixer(bd, &mA0, NC_, true); bind_mixer(bd, &mA1, ZW_, true);
        bind_mixer(bd, &mD0, NC_, true);
        break;
      case 3:  // params insertion order first, x_enc last
        bind_mixer(bd, &mA0, NC_, false); bind_mixer(bd, &mA1, ZW_, false);
        bind_mixer(bd, &mD0, NC_, false);
        bd.add(WGSZ, &Wg); bd.add(2, &bg); bd.add(XSZ, &xenc);
        break;
    }
    if ((int)bd.sz.size() != n_in) continue;
    bool ok = true;
    for (int i = 0; i < n_in; ++i) if (bd.sz[i] != (long)in_sizes[i]) { ok = false; break; }
    if (!ok) continue;
    for (int i = 0; i < n_in; ++i) *bd.dst[i] = (const float*)d_in[i];
    matched = true;
  }
  if (!matched) {  // best-effort fallback: combo 0 ordering
    Binding bd;
    bd.add(XSZ, &xenc);
    bind_mixer(bd, &mA0, NC_, false); bind_mixer(bd, &mA1, ZW_, false);
    bind_mixer(bd, &mD0, NC_, false);
    bd.add(WGSZ, &Wg); bd.add(2, &bg);
    int m = (int)bd.sz.size() < n_in ? (int)bd.sz.size() : n_in;
    for (int i = 0; i < m; ++i) *bd.dst[i] = (const float*)d_in[i];
  }

  // ---- workspace (bump allocator, fp32), ~408 MB total ----
  float* ws = (float*)d_ws;
  size_t off = 0;
  auto alloc = [&](size_t n) { float* p = ws + off; off += n; return p; };
  const size_t R = ROWS_;
  float* cA     = alloc(R * NC_);
  float* cD     = alloc(R * NC_);
  float* hid    = alloc(R * DM_);
  float* preds0 = alloc(R * OL_);
  float* wvec   = alloc(256);
  float* mixX   = alloc(R * DM_);
  float* tA     = alloc(R * DM_);
  float* tB     = alloc(R * 512);
  float* tC     = alloc(R * DM_);
  float* pd1a   = alloc(R * OL_);
  float* Z      = alloc(R * ZW_);
  float* preds1 = alloc(R * OL_);
  float* rec0   = cA;   // alias: cA dead after hid GEMM
  float* recD   = cD;   // alias: cD dead after packZ

  // ---- pipeline ----
  // 1) DWT (transpose fused): cA,cD (rows b*C+c, len 1027)
  dwt_kernel<<<dim3((NC_ + 3) / 4, BATCH_), 128, 0, stream>>>(xenc, cA, cD);
  // 2) hid = cA @ Wi0 + bi0   (M=16384, K=1027, N=128)
  gemm(stream, cA, NC_, mA0.Wi, DM_, mA0.bi, nullptr, 0, hid, DM_, DM_, NC_, 0);
  // 3) gating weights (hid rows per-batch are exactly h_ctx rows)
  gate_kernel<<<BATCH_, 256, 0, stream>>>(hid, Wg, bg, wvec);
  // 4) preds0 = hid @ Wo0 + bo0
  gemm(stream, hid, DM_, mA0.Wo, OL_, mA0.bo, nullptr, 0, preds0, OL_, OL_, DM_, 0);
  // 5) pd1a = mixer_cD(cD)
  run_mixer(stream, mD0, cD, NC_, NC_, pd1a, mixX, tA, tB, tC);
  // 6) rec0 = aconv(preds0)  (also the cA-recon base)
  recon_kernel<<<dim3(ROWS_, (PL_ + 127) / 128), 128, 0, stream>>>(preds0, nullptr, nullptr, rec0, PL_, 0);
  // 7) Z[:,0:1027]=cD, Z[:,1747:]=0
  packZ_kernel<<<dim3(ROWS_, (ZW_ + 255) / 256), 256, 0, stream>>>(cD, Z);
  // 8) Z[:,1027:1747] = rec0 + dconv(pd1a)   (the padded cA_recon)
  recon_kernel<<<dim3(ROWS_, (PL_ + 127) / 128), 128, 0, stream>>>(nullptr, pd1a, rec0, Z, ZW_, NC_);
  // 9) preds1 = mixer_cA1(Z)  (K=2054)
  run_mixer(stream, mA1, Z, ZW_, ZW_, preds1, mixX, tA, tB, tC);
  // 10) recD = dconv(preds1)
  recon_kernel<<<dim3(ROWS_, (PL_ + 127) / 128), 128, 0, stream>>>(nullptr, preds1, nullptr, recD, PL_, 0);
  // 11) out[b,o,c] = (w0+w1)*rec0 + w1*recD
  final_kernel<<<dim3((PL_ * CHAN_ + 255) / 256, BATCH_), 256, 0, stream>>>(rec0, recD, wvec, (float*)d_out);
}
